// MultiHeadAttention_30391188586812
// MI455X (gfx1250) — matrix operand order repaired and checked by reading, not compiled, not run on hardware
//
#include <hip/hip_runtime.h>
#include <hip/hip_bf16.h>
#include <math.h>

// Problem constants (reference: B=4, S=2048, E=1024, H=16, D=64)
#define B_   4
#define S_   2048
#define E_   1024
#define H_   16
#define D_   64
#define M_   (B_ * S_)        // 8192 tokens
#define SCALE_ 0.125f         // 1/sqrt(64)

typedef __attribute__((ext_vector_type(16))) __bf16 bf16x16;
typedef __attribute__((ext_vector_type(8)))  float  f32x8;
typedef __attribute__((ext_vector_type(4)))  int    v4i;

#define LDA   40   // GEMM LDS row pitch in halves (80B, 16B aligned)
#define PITCH 72   // fused-attn LDS row pitch in halves (144B, 16B aligned)

// Async global->LDS path (gfx1250 GLOBAL_LOAD_ASYNC_TO_LDS_B128, ASYNCcnt)
#if defined(__has_builtin)
#if __has_builtin(__builtin_amdgcn_global_load_async_to_lds_b128)
#define HAVE_ASYNC_LDS 1
#endif
#endif
#ifndef HAVE_ASYNC_LDS
#define HAVE_ASYNC_LDS 0
#endif

static __device__ __forceinline__ void wait_asynccnt0() {
#if defined(__has_builtin)
#if __has_builtin(__builtin_amdgcn_s_wait_asynccnt)
  __builtin_amdgcn_s_wait_asynccnt(0);
#define DID_WAIT_ASYNC 1
#endif
#endif
#ifndef DID_WAIT_ASYNC
  asm volatile("s_wait_asynccnt 0x0" ::: "memory");
#endif
}

// ---------------------------------------------------------------- helpers ---
static __device__ __forceinline__ __bf16 f2bf(float f) {
  union { float f; unsigned int u; } a; a.f = f;
  unsigned int u = a.u;
  u += 0x7fffu + ((u >> 16) & 1u);       // round-to-nearest-even
  union { unsigned short s; __bf16 b; } r;
  r.s = (unsigned short)(u >> 16);
  return r.b;
}

// A-matrix fragment (16x32 bf16): lanes 0-15 = row lr, K=0..15;
// lanes 16-31 hold K=16..31 (16 contiguous halves, matching frag_b).
static __device__ __forceinline__ bf16x16 frag_a(const __bf16* lds, int pitch,
                                                 int row, int k0, int kh) {
  const __bf16* p = lds + row * pitch + k0;
  bf16x16 r;
#pragma unroll
  for (int i = 0; i < 8; ++i) r[i]     = p[kh * 16 + i];
#pragma unroll
  for (int i = 0; i < 8; ++i) r[i + 8] = p[kh * 16 + 8 + i];
  return r;
}

// B-matrix fragment (32x16 bf16): lane = column; lanes 0-15 hold K=0..15,
// lanes 16-31 hold K=16..31 (16 contiguous halves).
static __device__ __forceinline__ bf16x16 frag_b(const __bf16* lds, int pitch,
                                                 int row, int k0, int kh) {
  const __bf16* p = lds + row * pitch + k0 + kh * 16;
  bf16x16 r;
#pragma unroll
  for (int i = 0; i < 16; ++i) r[i] = p[i];
  return r;
}

static __device__ __forceinline__ f32x8 wmma_bf16(bf16x16 a, bf16x16 b, f32x8 c) {
  return __builtin_amdgcn_wmma_f32_16x16x32_bf16(false, a, false, b, (short)0, c, false, false);
}

// Stage a 64x64 bf16 tile (row stride = D_ halves in global) into LDS.
// Uses async-to-LDS when available; caller must barrier afterwards.
static __device__ __forceinline__ void stage_64x64(const __bf16* __restrict__ g,
                                                   __bf16* l, int tid) {
#if HAVE_ASYNC_LDS
#pragma unroll
  for (int i = 0; i < 4; ++i) {
    int e = i * 128 + tid;                // 512 x 16B
    int row = e >> 3, q = e & 7;
    __builtin_amdgcn_global_load_async_to_lds_b128(
        (v4i __attribute__((address_space(1)))*)(g + (size_t)row * D_ + q * 8),
        (v4i __attribute__((address_space(3)))*)(l + row * PITCH + q * 8),
        0, 0);
  }
  wait_asynccnt0();
#else
#pragma unroll
  for (int i = 0; i < 4; ++i) {
    int e = i * 128 + tid;
    int row = e >> 3, q = e & 7;
    ((uint4*)(l + row * PITCH))[q] = ((const uint4*)(g + (size_t)row * D_))[q];
  }
#endif
}

// -------------------------------------------------- projection GEMM kernel ---
// Y = X @ W^T + bias.  X:[M,K=1024], W:[N=1024,K] fp32 row-major.
// OUT_MODE 0: bf16 [b,h,s,d]   (Q,K)
// OUT_MODE 1: bf16 [b,h,d,s]   (V transposed for attn@V)
// OUT_MODE 2: fp32 row-major [M,E] (final output projection)
template <bool IN_F32, int OUT_MODE>
__global__ void __launch_bounds__(128)
gemm_xwt(const float* __restrict__ Xf, const __bf16* __restrict__ Xb,
         const float* __restrict__ W, const float* __restrict__ bias,
         __bf16* __restrict__ Yb, float* __restrict__ Yf) {
  __shared__ __bf16 lA[64 * LDA];
  __shared__ __bf16 lB[64 * LDA];
  const int tid  = threadIdx.x;
  const int lane = tid & 31, wave = tid >> 5;
  const int wm = (wave >> 1) * 32, wn = (wave & 1) * 32;
  const int lr = lane & 15, kh = lane >> 4;
  const int tileM = blockIdx.x * 64, tileN = blockIdx.y * 64;
  const int K = E_;

  f32x8 acc[2][2] = {};
  for (int k0 = 0; k0 < K; k0 += 32) {
    if (k0 + 32 < K && tid < 64) {   // prefetch next K tile (global_prefetch_b8)
      __builtin_prefetch(W + (size_t)(tileN + tid) * K + k0 + 32, 0, 1);
      if constexpr (IN_F32) __builtin_prefetch(Xf + (size_t)(tileM + tid) * K + k0 + 32, 0, 1);
      else                  __builtin_prefetch(Xb + (size_t)(tileM + tid) * K + k0 + 32, 0, 1);
    }
    // ---- stage A tile (64x32) ----
    if constexpr (IN_F32) {
#pragma unroll
      for (int i = 0; i < 4; ++i) {        // 512 float4 groups
        int e = i * 128 + tid;
        int row = e >> 3, g = e & 7;
        const float4 xv = ((const float4*)(Xf + (size_t)(tileM + row) * K + k0))[g];
        union { __bf16 b[4]; uint2 u; } pk;
        pk.b[0] = f2bf(xv.x); pk.b[1] = f2bf(xv.y);
        pk.b[2] = f2bf(xv.z); pk.b[3] = f2bf(xv.w);
        ((uint2*)(lA + row * LDA))[g] = pk.u;
      }
    } else {
#pragma unroll
      for (int i = 0; i < 2; ++i) {        // 256 uint4 groups
        int e = i * 128 + tid;
        int row = e >> 2, q = e & 3;
        ((uint4*)(lA + row * LDA))[q] =
            ((const uint4*)(Xb + (size_t)(tileM + row) * K + k0))[q];
      }
    }
    // ---- stage B tile (64x32) from fp32 W ----
#pragma unroll
    for (int i = 0; i < 4; ++i) {
      int e = i * 128 + tid;
      int row = e >> 3, g = e & 7;
      const float4 wv = ((const float4*)(W + (size_t)(tileN + row) * K + k0))[g];
      union { __bf16 b[4]; uint2 u; } pw;
      pw.b[0] = f2bf(wv.x); pw.b[1] = f2bf(wv.y);
      pw.b[2] = f2bf(wv.z); pw.b[3] = f2bf(wv.w);
      ((uint2*)(lB + row * LDA))[g] = pw.u;
    }
    __syncthreads();
    bf16x16 a0 = frag_a(lA, LDA, wm + lr,      0, kh);
    bf16x16 a1 = frag_a(lA, LDA, wm + 16 + lr, 0, kh);
    bf16x16 b0 = frag_b(lB, LDA, wn + lr,      0, kh);
    bf16x16 b1 = frag_b(lB, LDA, wn + 16 + lr, 0, kh);
    acc[0][0] = wmma_bf16(a0, b0, acc[0][0]);
    acc[0][1] = wmma_bf16(a0, b1, acc[0][1]);
    acc[1][0] = wmma_bf16(a1, b0, acc[1][0]);
    acc[1][1] = wmma_bf16(a1, b1, acc[1][1]);
    __syncthreads();
  }

#pragma unroll
  for (int j = 0; j < 2; ++j) {
    int n = tileN + wn + j * 16 + lr;
    float bv = bias[n];
#pragma unroll
    for (int i = 0; i < 2; ++i) {
#pragma unroll
      for (int v = 0; v < 8; ++v) {
        int m = tileM + wm + i * 16 + v + kh * 8;
        float val = acc[i][j][v] + bv;
        if constexpr (OUT_MODE == 2) {
          Yf[(size_t)m * E_ + n] = val;
        } else {
          int b = m >> 11, s = m & (S_ - 1);
          int h = n >> 6,  d = n & (D_ - 1);
          size_t idx = (OUT_MODE == 0)
              ? ((((size_t)b * H_ + h) * S_ + s) * D_ + d)
              : ((((size_t)b * H_ + h) * D_ + d) * S_ + s);
          Yb[idx] = f2bf(val);
        }
      }
    }
  }
}

// ------------------------------------------- fused scores+softmax+attn@V ----
// Q:[bh,s,d] bf16, Kt:[bh,t,d] bf16, Vt:[bh,d,t] bf16.
// Writes normalized attn probs fp32 [bh,s,t] exactly once, and attended
// bf16 [b,s,h,d].  Each WG handles 64 s-rows of one head; each of the 4
// waves owns 16 rows x all 64 cols, so softmax row-reductions stay inside
// 16 lanes (__shfl_xor), no cross-wave reduction.
__global__ void __launch_bounds__(128)
fused_attn(const __bf16* __restrict__ Q, const __bf16* __restrict__ Kt,
           const __bf16* __restrict__ Vt, float* __restrict__ Attn,
           __bf16* __restrict__ Out) {
  __shared__ __bf16 lQ[64 * PITCH];
  __shared__ __bf16 lK[64 * PITCH];
  __shared__ __bf16 lP[64 * PITCH];
  const int tid = threadIdx.x, lane = tid & 31, wave = tid >> 5;
  const int lr = lane & 15, kh = lane >> 4;
  const int wm = wave * 16;                 // wave's 16-row slice
  const int blkS = blockIdx.x * 64, bh = blockIdx.y;
  const __bf16* Qh = Q  + ((size_t)bh * S_ + blkS) * D_;
  const __bf16* Kh = Kt + (size_t)bh * S_ * D_;
  const __bf16* Vh = Vt + (size_t)bh * D_ * S_;
  float* attn_h = Attn + (size_t)bh * S_ * S_;

  // Stage the 64x64 Q block once (resident for the whole kernel).
  stage_64x64(Qh, lQ, tid);

  float m[8], l[8];
#pragma unroll
  for (int v = 0; v < 8; ++v) { m[v] = -3.4e38f; l[v] = 0.f; }

  // ---- Phase A: online row max / sum-exp (scores never stored) ----
  for (int t0 = 0; t0 < S_; t0 += 64) {
    if (t0 + 64 < S_ && tid < 64)
      __builtin_prefetch(Kh + (size_t)(t0 + 64 + tid) * D_, 0, 1);
    stage_64x64(Kh + (size_t)t0 * D_, lK, tid);
    __syncthreads();
    f32x8 sc[4] = {};
#pragma unroll
    for (int d0 = 0; d0 < 64; d0 += 32) {
      bf16x16 a = frag_a(lQ, PITCH, wm + lr, d0, kh);
#pragma unroll
      for (int j = 0; j < 4; ++j) {
        bf16x16 b = frag_b(lK, PITCH, j * 16 + lr, d0, kh);
        sc[j] = wmma_bf16(a, b, sc[j]);
      }
    }
#pragma unroll
    for (int v = 0; v < 8; ++v) {
      float tm = -3.4e38f;
#pragma unroll
      for (int j = 0; j < 4; ++j) tm = fmaxf(tm, sc[j][v]);
      tm *= SCALE_;
#pragma unroll
      for (int off = 1; off < 16; off <<= 1) tm = fmaxf(tm, __shfl_xor(tm, off, 32));
      float nm = fmaxf(m[v], tm);
      float es = 0.f;
#pragma unroll
      for (int j = 0; j < 4; ++j) es += __expf(sc[j][v] * SCALE_ - nm);
#pragma unroll
      for (int off = 1; off < 16; off <<= 1) es += __shfl_xor(es, off, 32);
      l[v] = l[v] * __expf(m[v] - nm) + es;
      m[v] = nm;
    }
    __syncthreads();
  }

  float il[8];
#pragma unroll
  for (int v = 0; v < 8; ++v) il[v] = 1.0f / l[v];

  // ---- Phase B: recompute scores, emit normalized probs once, P @ V ----
  f32x8 oAcc[4] = {};
  for (int t0 = 0; t0 < S_; t0 += 64) {
    if (t0 + 64 < S_ && tid < 64)
      __builtin_prefetch(Kh + (size_t)(t0 + 64 + tid) * D_, 0, 1);
    stage_64x64(Kh + (size_t)t0 * D_, lK, tid);
    __syncthreads();
    f32x8 sc[4] = {};
#pragma unroll
    for (int d0 = 0; d0 < 64; d0 += 32) {
      bf16x16 a = frag_a(lQ, PITCH, wm + lr, d0, kh);
#pragma unroll
      for (int j = 0; j < 4; ++j) {
        bf16x16 b = frag_b(lK, PITCH, j * 16 + lr, d0, kh);
        sc[j] = wmma_bf16(a, b, sc[j]);
      }
    }
#pragma unroll
    for (int j = 0; j < 4; ++j) {
      int tcol = t0 + j * 16 + lr;
#pragma unroll
      for (int v = 0; v < 8; ++v) {
        int srow = wm + kh * 8 + v;
        float p = __expf(sc[j][v] * SCALE_ - m[v]) * il[v];
        attn_h[(size_t)(blkS + srow) * S_ + tcol] = p;       // the 1GB output
        lP[srow * PITCH + j * 16 + lr] = f2bf(p);
      }
    }
    __syncthreads();
#pragma unroll
    for (int tt = 0; tt < 64; tt += 32) {
      bf16x16 a = frag_a(lP, PITCH, wm + lr, tt, kh);
#pragma unroll
      for (int j = 0; j < 4; ++j) {
        const __bf16* g = Vh + (size_t)(j * 16 + lr) * S_ + t0 + tt + kh * 16;
        bf16x16 b;
#pragma unroll
        for (int q = 0; q < 16; ++q) b[q] = g[q];            // V: L2-resident
        oAcc[j] = wmma_bf16(a, b, oAcc[j]);
      }
    }
    __syncthreads();
  }

  const int b = bh >> 4, h = bh & 15;
#pragma unroll
  for (int j = 0; j < 4; ++j) {
    int d = j * 16 + lr;
#pragma unroll
    for (int v = 0; v < 8; ++v) {
      int s = blkS + wm + kh * 8 + v;
      Out[(((size_t)b * S_ + s) * H_ + h) * D_ + d] = f2bf(oAcc[j][v]);
    }
  }
}

// ------------------------------------------------------------------ launch ---
extern "C" void kernel_launch(void* const* d_in, const int* in_sizes, int n_in,
                              void* d_out, int out_size, void* d_ws, size_t ws_size,
                              hipStream_t stream) {
  (void)in_sizes; (void)n_in; (void)out_size; (void)ws_size;
  const float* query = (const float*)d_in[0];
  const float* key_t = (const float*)d_in[1];
  const float* value = (const float*)d_in[2];
  const float* Wq = (const float*)d_in[3];  const float* bq = (const float*)d_in[4];
  const float* Wk = (const float*)d_in[5];  const float* bk = (const float*)d_in[6];
  const float* Wv = (const float*)d_in[7];  const float* bv = (const float*)d_in[8];
  const float* Wo = (const float*)d_in[9];  const float* bo = (const float*)d_in[10];

  float* out  = (float*)d_out;                    // [B,S,E] fp32
  float* attn = out + (size_t)M_ * E_;            // [B,H,S,S] fp32

  __bf16* Qb  = (__bf16*)d_ws;                    // [b,h,s,d]
  __bf16* Kb  = Qb  + (size_t)M_ * E_;            // [b,h,t,d]
  __bf16* Vtb = Kb  + (size_t)M_ * E_;            // [b,h,d,t]
  __bf16* Att = Vtb + (size_t)M_ * E_;            // [b,s,h,d] == [M,E]

  dim3 gP(M_ / 64, E_ / 64);                      // 128 x 16
  gemm_xwt<true, 0><<<gP, 128, 0, stream>>>(query, nullptr, Wq, bq, Qb,  nullptr);
  gemm_xwt<true, 0><<<gP, 128, 0, stream>>>(key_t, nullptr, Wk, bk, Kb,  nullptr);
  gemm_xwt<true, 1><<<gP, 128, 0, stream>>>(value, nullptr, Wv, bv, Vtb, nullptr);

  fused_attn<<<dim3(S_ / 64, B_ * H_), 128, 0, stream>>>(Qb, Kb, Vtb, attn, Att);

  gemm_xwt<false, 2><<<gP, 128, 0, stream>>>(nullptr, Att, Wo, bo, nullptr, out);
}
